// transformerDecoder_15599321219549
// MI455X (gfx1250) — matrix operand order repaired and checked by reading, not compiled, not run on hardware
//
#include <hip/hip_runtime.h>
#include <hip/hip_bf16.h>
#include <cmath>

// ---------------------------------------------------------------------------
// GPT-style decoder forward on gfx1250 via v_wmma_f32_16x16x32_f16 with
// async-to-LDS double-buffered B tiles (ASYNCcnt path).
// B=2, T=1024, D=1024, L=4, H=8, HD=128, V=32000.
// ---------------------------------------------------------------------------

typedef _Float16 f16;
typedef __attribute__((ext_vector_type(8)))  _Float16 v8h;
typedef __attribute__((ext_vector_type(16))) _Float16 v16h;
typedef __attribute__((ext_vector_type(8)))  float    v8f;

#define CB 2
#define CT 1024
#define CD 1024
#define CL 4
#define CH 8
#define CHD 128
#define CV 32000
#define CM (CB * CT)      // 2048 rows
#define CD3 (3 * CD)
#define CD4 (4 * CD)
#define CBH (CB * CH)     // 16

#define BLK_M 128         // per workgroup (4 waves x 2 slabs x 16)
#define BLK_N 64
#define KSTEP 32

// ---------------------------------------------------------------------------
// Workgroup-level WMMA GEMM:  C[z] = act(scale * A[z] * Bt[z]^T + bias) + resid
//   A  : M x K  f16 row-major        (per-z stride sAz elements)
//   Bt : N x K  f16 row-major        (B transposed; per-z stride sBz)
//   C  : M x N  f32 or f16 row-major (per-z stride sCz)
// Block = 128 threads = 4 waves. Each wave owns 2 M-slabs (16 rows each) and
// the full 64-wide N strip -> 8 WMMA accumulators. The shared 64x32 B tile is
// staged in LDS via global_load_async_to_lds_b128 with double buffering.
// A/B fragments use contiguous k (k = 16*hi + e); f32 C/D follows CDNA5 ISA 7.12.2 layout.
// ---------------------------------------------------------------------------
__global__ void __launch_bounds__(128)
gemm_f16_wmma(const _Float16* __restrict__ A,
              const _Float16* __restrict__ Bt,
              const float* __restrict__ bias,
              const float* __restrict__ resid,
              void* __restrict__ Cout,
              int M, int N, int K,
              long long sAz, long long sBz, long long sCz,
              float scale, int causal, int gelu, int outF16)
{
    __shared__ __align__(64) _Float16 bsh[2][BLK_N * KSTEP]; // 2 x 4KB

    const int tid  = threadIdx.x;
    const int lane = tid & 31;
    const int wave = tid >> 5;
    const int hi   = lane >> 4;
    const int lm   = lane & 15;
    const int n0   = blockIdx.x * BLK_N;
    const int m0   = blockIdx.y * BLK_M;
    const int z    = blockIdx.z;

    const _Float16* Az = A  + (size_t)z * sAz;
    const _Float16* Bz = Bt + (size_t)z * sBz;

    // -------- causal fast path: block entirely above the diagonal ----------
    if (causal && n0 > m0 + BLK_M - 1) {
        for (int s = 0; s < 2; ++s)
            for (int j = 0; j < 4; ++j)
                for (int r = 0; r < 8; ++r) {
                    const int row = m0 + wave * 32 + s * 16 + hi * 8 + r;
                    const int col = n0 + j * 16 + lm;
                    const size_t idx = (size_t)z * sCz + (size_t)row * N + col;
                    ((float*)Cout)[idx] = -1e30f;
                }
        return;
    }

    // -------- async B-tile loader: 64 rows x 32 halfs = 256 16B chunks -----
    // chunk q (0..255): row = q>>2, c = q&3; each thread issues q=tid, tid+128
    const uint32_t lds_base = (uint32_t)(size_t)&bsh[0][0];
    auto issueB = [&](int buf, int k0) {
        #pragma unroll
        for (int r = 0; r < 2; ++r) {
            const int q   = tid + r * 128;
            const int row = q >> 2;
            const int c   = q & 3;
            const uint64_t ga = (uint64_t)(const char*)(Bz + (size_t)(n0 + row) * K + k0 + c * 8);
            const uint32_t lo = lds_base + (uint32_t)buf * (BLK_N * KSTEP * 2) + (uint32_t)q * 16;
            asm volatile("global_load_async_to_lds_b128 %0, %1, off"
                         :: "v"(lo), "v"(ga) : "memory");
        }
    };

    // A fragment rows for this wave's two M-slabs
    const _Float16* arow0 = Az + (size_t)(m0 + wave * 32 + lm) * K;
    const _Float16* arow1 = arow0 + (size_t)16 * K;

    v8f acc[2][4];
    #pragma unroll
    for (int s = 0; s < 2; ++s)
        #pragma unroll
        for (int j = 0; j < 4; ++j) acc[s][j] = v8f{};

    issueB(0, 0);

    for (int k0 = 0, it = 0; k0 < K; k0 += KSTEP, ++it) {
        const int cur = it & 1;
        if (k0 + KSTEP < K) {
            issueB(1 - cur, k0 + KSTEP);
            // let the 2 just-issued loads stay in flight; wait for current tile
            asm volatile("s_wait_asynccnt 0x2" ::: "memory");
        } else {
            asm volatile("s_wait_asynccnt 0x0" ::: "memory");
        }
        __syncthreads();   // all waves' async writes for `cur` visible

        // A fragments (lanes<16 hold K 0..15; lanes>=16 hold 16..31)
        v8h a0lo = *(const v8h*)(arow0 + k0 + hi * 16);
        v8h a0hi = *(const v8h*)(arow0 + k0 + hi * 16 + 8);
        v8h a1lo = *(const v8h*)(arow1 + k0 + hi * 16);
        v8h a1hi = *(const v8h*)(arow1 + k0 + hi * 16 + 8);
        v16h a0 = __builtin_shufflevector(a0lo, a0hi, 0,1,2,3,4,5,6,7,8,9,10,11,12,13,14,15);
        v16h a1 = __builtin_shufflevector(a1lo, a1hi, 0,1,2,3,4,5,6,7,8,9,10,11,12,13,14,15);

        // B fragments from LDS: lane holds column n = j*16+lm, halves K = hi*16+e
        const _Float16* bb = &bsh[cur][0];
        #pragma unroll
        for (int j = 0; j < 4; ++j) {
            v16h b = *(const v16h*)(bb + (size_t)(j * 16 + lm) * KSTEP + hi * 16);
            acc[0][j] = __builtin_amdgcn_wmma_f32_16x16x32_f16(false, a0, false, b, (short)0, acc[0][j], false, false);
            acc[1][j] = __builtin_amdgcn_wmma_f32_16x16x32_f16(false, a1, false, b, (short)0, acc[1][j], false, false);
        }
        __syncthreads();   // all waves done reading `cur` before it is rewritten
    }

    // -------- epilogue: C/D layout row = base + hi*8 + r, col = n-tile + lm --
    for (int s = 0; s < 2; ++s) {
        const int mbase = m0 + wave * 32 + s * 16 + hi * 8;
        for (int j = 0; j < 4; ++j) {
            const int col = n0 + j * 16 + lm;
            const float bval = bias ? bias[col] : 0.0f;
            for (int r = 0; r < 8; ++r) {
                const int row = mbase + r;
                float v = acc[s][j][r] * scale + bval;
                if (causal && col > row) v = -1e30f;
                if (gelu)  v = 0.5f * v * (1.0f + erff(v * 0.70710678118654752f));
                const size_t idx = (size_t)z * sCz + (size_t)row * N + col;
                if (resid) v += resid[idx];
                if (outF16) ((_Float16*)Cout)[idx] = (_Float16)v;
                else        ((float*)Cout)[idx]    = v;
            }
        }
    }
}

// ---------------------------------------------------------------------------
// LayerNorm (one 256-thread block per row), fp32 in -> f16 out (GEMM A operand)
// ---------------------------------------------------------------------------
__global__ void __launch_bounds__(256)
layernorm_to_f16(const float* __restrict__ X, const float* __restrict__ g,
                 const float* __restrict__ b, _Float16* __restrict__ Y, int D)
{
    const int row = blockIdx.x;
    const int tid = threadIdx.x;
    const float* x = X + (size_t)row * D;
    __shared__ float s1[256], s2[256];

    float a = 0.0f, sq = 0.0f;
    for (int i = tid; i < D; i += 256) { float v = x[i]; a += v; sq += v * v; }
    s1[tid] = a; s2[tid] = sq;
    __syncthreads();
    for (int off = 128; off > 0; off >>= 1) {
        if (tid < off) { s1[tid] += s1[tid + off]; s2[tid] += s2[tid + off]; }
        __syncthreads();
    }
    const float mean = s1[0] / D;
    const float var  = s2[0] / D - mean * mean;
    const float rstd = rsqrtf(var + 1e-5f);

    _Float16* y = Y + (size_t)row * D;
    for (int i = tid; i < D; i += 256)
        y[i] = (_Float16)((x[i] - mean) * rstd * g[i] + b[i]);
}

// ---------------------------------------------------------------------------
// Row softmax over masked scores (mask already applied as -1e30), f16 out
// ---------------------------------------------------------------------------
__global__ void __launch_bounds__(256)
softmax_rows_f16(const float* __restrict__ S, _Float16* __restrict__ A, int Tlen)
{
    const int row = blockIdx.x;
    const int tid = threadIdx.x;
    const float* s = S + (size_t)row * Tlen;
    __shared__ float red[256];

    float mx = -1e30f;
    for (int i = tid; i < Tlen; i += 256) mx = fmaxf(mx, s[i]);
    red[tid] = mx;
    __syncthreads();
    for (int off = 128; off > 0; off >>= 1) {
        if (tid < off) red[tid] = fmaxf(red[tid], red[tid + off]);
        __syncthreads();
    }
    mx = red[0];
    __syncthreads();

    float sum = 0.0f;
    for (int i = tid; i < Tlen; i += 256) sum += __expf(s[i] - mx);
    red[tid] = sum;
    __syncthreads();
    for (int off = 128; off > 0; off >>= 1) {
        if (tid < off) red[tid] += red[tid + off];
        __syncthreads();
    }
    const float inv = 1.0f / red[0];

    _Float16* a = A + (size_t)row * Tlen;
    for (int i = tid; i < Tlen; i += 256)
        a[i] = (_Float16)(__expf(s[i] - mx) * inv);
}

// ---------------------------------------------------------------------------
// Embedding: h[b*T+t][d] = tok_emb[x[b*T+t]][d] + pos_emb[t][d]
// ---------------------------------------------------------------------------
__global__ void embed_kernel(const int* __restrict__ x,
                             const float* __restrict__ tok,
                             const float* __restrict__ pos,
                             float* __restrict__ h)
{
    const int i = blockIdx.x * blockDim.x + threadIdx.x;
    if (i >= CM * CD) return;
    const int rg = i / CD;          // b*T + t
    const int d  = i % CD;
    const int t  = rg % CT;
    const int id = x[rg];
    h[i] = tok[(size_t)id * CD + d] + pos[(size_t)t * CD + d];
}

// ---------------------------------------------------------------------------
// Split QKV (fp32 [M][3D]) into Qh/Kh f16 [BH][T][HD] and Vt f16 [BH][HD][T]
// ---------------------------------------------------------------------------
__global__ void split_qkv_kernel(const float* __restrict__ qkv,
                                 _Float16* __restrict__ Qh,
                                 _Float16* __restrict__ Kh,
                                 _Float16* __restrict__ Vt)
{
    const int i = blockIdx.x * blockDim.x + threadIdx.x;
    if (i >= CBH * CT * CHD) return;
    const int bh  = i / (CT * CHD);
    const int rem = i % (CT * CHD);
    const int t   = rem / CHD;
    const int d   = rem % CHD;
    const int b   = bh >> 3;        // /H
    const int hh  = bh & 7;         // %H
    const size_t rowoff = (size_t)(b * CT + t) * CD3;
    const int col = hh * CHD + d;
    Qh[i] = (_Float16)qkv[rowoff + col];
    Kh[i] = (_Float16)qkv[rowoff + CD + col];
    Vt[(size_t)bh * CHD * CT + (size_t)d * CT + t] = (_Float16)qkv[rowoff + 2 * CD + col];
}

// ---------------------------------------------------------------------------
// Merge heads: [BH][T][HD] f16 -> [M][D] f16
// ---------------------------------------------------------------------------
__global__ void merge_heads_kernel(const _Float16* __restrict__ Oh,
                                   _Float16* __restrict__ Om)
{
    const int i = blockIdx.x * blockDim.x + threadIdx.x;
    if (i >= CM * CD) return;
    const int row = i / CD;         // b*T + t
    const int c   = i % CD;
    const int hh  = c / CHD;
    const int d   = c % CHD;
    const int b   = row / CT;
    const int t   = row % CT;
    Om[i] = Oh[((size_t)(b * CH + hh) * CT + t) * CHD + d];
}

// ---------------------------------------------------------------------------
// Weight convert + transpose: W f32 [K][N] -> Wt f16 [N][K]
// ---------------------------------------------------------------------------
__global__ void convert_wT_kernel(const float* __restrict__ W,
                                  _Float16* __restrict__ Wt, int K, int N)
{
    const int i = blockIdx.x * blockDim.x + threadIdx.x;
    if (i >= N * K) return;
    const int n = i / K;
    const int k = i % K;
    Wt[i] = (_Float16)W[(size_t)k * N + n];
}

// ---------------------------------------------------------------------------
extern "C" void kernel_launch(void* const* d_in, const int* in_sizes, int n_in,
                              void* d_out, int out_size, void* d_ws, size_t ws_size,
                              hipStream_t stream)
{
    (void)in_sizes; (void)n_in; (void)out_size; (void)ws_size;

    const int*   x       = (const int*)  d_in[0];
    const float* tok_emb = (const float*)d_in[1];
    const float* pos_emb = (const float*)d_in[2];
    const float* ln1_g   = (const float*)d_in[3];
    const float* ln1_b   = (const float*)d_in[4];
    const float* wqkv    = (const float*)d_in[5];
    const float* bqkv    = (const float*)d_in[6];
    const float* wo      = (const float*)d_in[7];
    const float* bo      = (const float*)d_in[8];
    const float* ln2_g   = (const float*)d_in[9];
    const float* ln2_b   = (const float*)d_in[10];
    const float* wfc     = (const float*)d_in[11];
    const float* wfc2    = (const float*)d_in[12];
    const float* fn_g    = (const float*)d_in[13];
    const float* fn_b    = (const float*)d_in[14];
    const float* wout    = (const float*)d_in[15];
    const float* bout    = (const float*)d_in[16];
    float* out = (float*)d_out;

    // ---- workspace carve-up (256B aligned) ----
    char* wsb = (char*)d_ws;
    size_t off = 0;
    auto take = [&](size_t bytes) -> char* {
        char* p = wsb + off;
        off = (off + bytes + 255) & ~(size_t)255;
        return p;
    };
    float* h       = (float*)take((size_t)CM * CD * 4);
    f16*   xn16    = (f16*)  take((size_t)CM * CD * 2);
    float* qkv     = (float*)take((size_t)CM * CD3 * 4);
    f16*   Qh      = (f16*)  take((size_t)CBH * CT * CHD * 2);
    f16*   Kh      = (f16*)  take((size_t)CBH * CT * CHD * 2);
    f16*   Vt      = (f16*)  take((size_t)CBH * CHD * CT * 2);
    float* scores  = (float*)take((size_t)CBH * CT * CT * 4);
    f16*   att16   = (f16*)  take((size_t)CBH * CT * CT * 2);
    f16*   outh16  = (f16*)  take((size_t)CBH * CT * CHD * 2);
    f16*   attnm16 = (f16*)  take((size_t)CM * CD * 2);
    f16*   mlp16   = (f16*)  take((size_t)CM * CD4 * 2);
    f16*   wqkvT   = (f16*)  take((size_t)CL * CD * CD3 * 2);
    f16*   woT     = (f16*)  take((size_t)CL * CD * CD * 2);
    f16*   wfcT    = (f16*)  take((size_t)CL * CD * CD4 * 2);
    f16*   wfc2T   = (f16*)  take((size_t)CL * CD4 * CD * 2);
    f16*   woutT   = (f16*)  take((size_t)CD * CV * 2);

    auto gemm = [&](const f16* A, const f16* Bt, const float* bias,
                    const float* resid, void* C, int M, int N, int K,
                    long long sA, long long sB, long long sC, int Z,
                    float scale, int causal, int gelu, int outF16) {
        dim3 grid(N / BLK_N, M / BLK_M, Z);
        hipLaunchKernelGGL(gemm_f16_wmma, grid, dim3(128), 0, stream,
                           A, Bt, bias, resid, C, M, N, K, sA, sB, sC,
                           scale, causal, gelu, outF16);
    };
    auto conv = [&](const float* W, f16* Wt, int K, int N) {
        int n = N * K;
        hipLaunchKernelGGL(convert_wT_kernel, dim3((n + 255) / 256), dim3(256),
                           0, stream, W, Wt, K, N);
    };

    // ---- weight convert+transpose (fp32 [K][N] -> f16 [N][K]) ----
    for (int l = 0; l < CL; ++l) {
        conv(wqkv + (size_t)l * CD * CD3, wqkvT + (size_t)l * CD3 * CD, CD, CD3);
        conv(wo   + (size_t)l * CD * CD,  woT   + (size_t)l * CD  * CD, CD, CD);
        conv(wfc  + (size_t)l * CD * CD4, wfcT  + (size_t)l * CD4 * CD, CD, CD4);
        conv(wfc2 + (size_t)l * CD4 * CD, wfc2T + (size_t)l * CD  * CD4, CD4, CD);
    }
    conv(wout, woutT, CD, CV);

    // ---- embedding ----
    hipLaunchKernelGGL(embed_kernel, dim3((CM * CD + 255) / 256), dim3(256),
                       0, stream, x, tok_emb, pos_emb, h);

    const float inv_sqrt_hd = 0.08838834764831845f;  // 1/sqrt(128)

    // ---- transformer layers ----
    for (int l = 0; l < CL; ++l) {
        // LN1 -> f16
        hipLaunchKernelGGL(layernorm_to_f16, dim3(CM), dim3(256), 0, stream,
                           h, ln1_g + l * CD, ln1_b + l * CD, xn16, CD);
        // QKV projection (+bias)
        gemm(xn16, wqkvT + (size_t)l * CD3 * CD, bqkv + l * CD3, nullptr, qkv,
             CM, CD3, CD, 0, 0, 0, 1, 1.0f, 0, 0, 0);
        // split heads (V transposed for the att*V GEMM B-operand)
        hipLaunchKernelGGL(split_qkv_kernel, dim3((CBH * CT * CHD + 255) / 256),
                           dim3(256), 0, stream, qkv, Qh, Kh, Vt);
        // scores = Q K^T / sqrt(hd), causal mask in epilogue
        gemm(Qh, Kh, nullptr, nullptr, scores, CT, CT, CHD,
             (long long)CT * CHD, (long long)CT * CHD, (long long)CT * CT,
             CBH, inv_sqrt_hd, 1, 0, 0);
        // softmax rows -> f16
        hipLaunchKernelGGL(softmax_rows_f16, dim3(CBH * CT), dim3(256), 0, stream,
                           scores, att16, CT);
        // out = att @ V
        gemm(att16, Vt, nullptr, nullptr, outh16, CT, CHD, CT,
             (long long)CT * CT, (long long)CHD * CT, (long long)CT * CHD,
             CBH, 1.0f, 0, 0, 1);
        // merge heads
        hipLaunchKernelGGL(merge_heads_kernel, dim3((CM * CD + 255) / 256),
                           dim3(256), 0, stream, outh16, attnm16);
        // output projection + bias + residual (in place on h)
        gemm(attnm16, woT + (size_t)l * CD * CD, bo + l * CD, h, h,
             CM, CD, CD, 0, 0, 0, 1, 1.0f, 0, 0, 0);
        // LN2 -> f16
        hipLaunchKernelGGL(layernorm_to_f16, dim3(CM), dim3(256), 0, stream,
                           h, ln2_g + l * CD, ln2_b + l * CD, xn16, CD);
        // MLP fc1 (no bias) + exact GELU -> f16
        gemm(xn16, wfcT + (size_t)l * CD4 * CD, nullptr, nullptr, mlp16,
             CM, CD4, CD, 0, 0, 0, 1, 1.0f, 0, 1, 1);
        // MLP fc2 (no bias) + residual (in place on h)
        gemm(mlp16, wfc2T + (size_t)l * CD * CD4, nullptr, h, h,
             CM, CD, CD4, 0, 0, 0, 1, 1.0f, 0, 0, 0);
    }

    // ---- final LN + vocab head ----
    hipLaunchKernelGGL(layernorm_to_f16, dim3(CM), dim3(256), 0, stream,
                       h, fn_g, fn_b, xn16, CD);
    gemm(xn16, woutT, bout, nullptr, out, CM, CV, CD, 0, 0, 0, 1,
         1.0f, 0, 0, 0);
}
